// ModelSbj_66786741453111
// MI455X (gfx1250) — compile-verified, matrix operand order checked by reading, not run on hardware
//
#include <hip/hip_runtime.h>
#include <hip/hip_bf16.h>

typedef _Float16 f16;
typedef __attribute__((ext_vector_type(16))) _Float16 v16h;
typedef __attribute__((ext_vector_type(8)))  _Float16 v8h;
typedef __attribute__((ext_vector_type(8)))  float    v8f;

#define T_    256
#define B_    64
#define D_    300
#define DP_   320
#define H_    256
#define G4_   1024
#define K_    37
#define ROWS_ (T_*B_)   // 16384

__device__ __forceinline__ float sigmf(float x){ return 1.0f/(1.0f+__expf(-x)); }

// ---------------- weight convert: f32 [N][K] -> f16 [N][Kpad] (zero pad) ----
__global__ void k_cvt(const float* __restrict__ src, f16* __restrict__ dst,
                      int N, int K, int Kpad){
  int idx = blockIdx.x*blockDim.x + threadIdx.x;
  if (idx >= N*Kpad) return;
  int n = idx / Kpad, k = idx - n*Kpad;
  dst[idx] = (k < K) ? (f16)src[(size_t)n*K + k] : (f16)0.0f;
}

// ---------------- embedding gather: x[t*B+b][0..DP) f16 ---------------------
__global__ void k_gather(const int* __restrict__ text, const float* __restrict__ emb,
                         f16* __restrict__ x){
  int row = blockIdx.x;           // t*B + b
  int t = row >> 6, b = row & 63;
  int tok = text[b*T_ + t];
  const float* e = emb + (size_t)tok * D_;
  f16* xr = x + (size_t)row * DP_;
  for (int c = threadIdx.x; c < DP_; c += blockDim.x)
    xr[c] = (c < D_) ? (f16)e[c] : (f16)0.0f;
}

// ---------------- big input-gate GEMM: C[16384][1024] = A f16 * W^T + bias --
// per wave: one 16-row M tile x 64 cols (4 N tiles), k-loop over Kpad/32
__global__ __launch_bounds__(256)
void k_gemm_ih(const f16* __restrict__ A, const f16* __restrict__ W,
               const float* __restrict__ bias, float* __restrict__ C, int Kpad){
  int lane  = threadIdx.x & 31;
  int wv    = blockIdx.x * 8 + (threadIdx.x >> 5);   // 0..16383
  int mt    = wv >> 4;                               // 0..1023
  int ng    = wv & 15;                               // 0..15
  int l15   = lane & 15;
  int base8 = (lane >= 16) ? 8 : 0;
  const f16* Arow = A + (size_t)(mt*16 + l15) * Kpad;

  v8f z = {};
  v8f acc[4] = {z, z, z, z};
  int nkt = Kpad >> 5;
  for (int kt = 0; kt < nkt; kt++){
    int k0 = kt*32 + base8;
    v16h af;
    { v8h lo = *(const v8h*)(Arow + k0);
      v8h hi = *(const v8h*)(Arow + k0 + 16);
      #pragma unroll
      for (int q = 0; q < 8; q++){ af[q] = lo[q]; af[q+8] = hi[q]; } }
    #pragma unroll
    for (int i = 0; i < 4; i++){
      const f16* Wr = W + (size_t)(ng*64 + i*16 + l15) * Kpad + k0;
      v16h bf;
      v8h lo = *(const v8h*)Wr;
      v8h hi = *(const v8h*)(Wr + 16);
      #pragma unroll
      for (int q = 0; q < 8; q++){ bf[q] = lo[q]; bf[q+8] = hi[q]; }
      acc[i] = __builtin_amdgcn_wmma_f32_16x16x32_f16(false, af, false, bf,
                                                      (short)0, acc[i], false, false);
    }
  }
  int mrow = base8;  // 0 or 8
  #pragma unroll
  for (int i = 0; i < 4; i++){
    int ncol = ng*64 + i*16 + l15;
    float bv = bias[ncol];
    #pragma unroll
    for (int v = 0; v < 8; v++){
      int r = mt*16 + mrow + v;
      C[(size_t)r * G4_ + ncol] = acc[i][v] + bv;
    }
  }
}

// ---------------- persistent LSTM recurrence (one WG per direction) ---------
// 512 threads = 16 waves. Wave w owns h-cols [16w,16w+16) and the 4 gate
// column blocks {j, j+256, j+512, j+768} -> gate combine + c state stay
// wave-local in registers (v8f frags matching WMMA C/D layout).
__global__ __launch_bounds__(512)
void k_lstm(const float* __restrict__ Gf, const float* __restrict__ Gb,
            const f16* __restrict__ Whf, const f16* __restrict__ Whb,
            f16* __restrict__ hout){
  const float* G  = (blockIdx.x == 0) ? Gf : Gb;
  const f16*  Wh  = (blockIdx.x == 0) ? Whf : Whb;
  int dirOff      = (blockIdx.x == 0) ? 0 : H_;
  int rev         = blockIdx.x;

  __shared__ f16 hsh[B_ * H_];                 // 32 KB
  int lane  = threadIdx.x & 31;
  int wv    = threadIdx.x >> 5;                // 0..15
  int l15   = lane & 15;
  int base8 = (lane >= 16) ? 8 : 0;

  for (int i = threadIdx.x; i < B_*H_; i += blockDim.x) hsh[i] = (f16)0.0f;
  v8f z = {};
  v8f cfr[4] = {z, z, z, z};
  __syncthreads();

  for (int t = 0; t < T_; t++){
    int te = rev ? (T_-1-t) : t;
    const float* Grow = G + (size_t)te * B_ * G4_;
    v8f acc[16];
    #pragma unroll
    for (int i = 0; i < 16; i++) acc[i] = z;

    for (int kt = 0; kt < 8; kt++){
      int k0 = kt*32 + base8;
      v16h af[4];
      #pragma unroll
      for (int mt = 0; mt < 4; mt++){
        const f16* hp = hsh + (mt*16 + l15)*H_ + k0;
        v8h lo = *(const v8h*)hp;
        v8h hi = *(const v8h*)(hp + 16);
        #pragma unroll
        for (int q = 0; q < 8; q++){ af[mt][q] = lo[q]; af[mt][q+8] = hi[q]; }
      }
      #pragma unroll
      for (int g = 0; g < 4; g++){
        const f16* wp = Wh + (size_t)(g*H_ + wv*16 + l15) * H_ + k0;
        v16h bf;
        v8h lo = *(const v8h*)wp;
        v8h hi = *(const v8h*)(wp + 16);
        #pragma unroll
        for (int q = 0; q < 8; q++){ bf[q] = lo[q]; bf[q+8] = hi[q]; }
        #pragma unroll
        for (int mt = 0; mt < 4; mt++)
          acc[g*4+mt] = __builtin_amdgcn_wmma_f32_16x16x32_f16(false, af[mt], false, bf,
                                                     (short)0, acc[g*4+mt], false, false);
      }
    }
    __syncthreads();   // all reads of hsh done before overwriting

    int j = wv*16 + l15;
    #pragma unroll
    for (int mt = 0; mt < 4; mt++){
      #pragma unroll
      for (int v = 0; v < 8; v++){
        int r = mt*16 + base8 + v;
        const float* gr = Grow + (size_t)r * G4_;
        float zi = acc[0*4+mt][v] + gr[j];
        float zf = acc[1*4+mt][v] + gr[H_   + j];
        float zg = acc[2*4+mt][v] + gr[2*H_ + j];
        float zo = acc[3*4+mt][v] + gr[3*H_ + j];
        float c  = sigmf(zf)*cfr[mt][v] + sigmf(zi)*tanhf(zg);
        cfr[mt][v] = c;
        float h  = sigmf(zo)*tanhf(c);
        hsh[r*H_ + j] = (f16)h;
        hout[((size_t)te*B_ + r)*(2*H_) + dirOff + j] = (f16)h;
      }
    }
    __syncthreads();
  }
}

// ---------------- emissions: em[row][k] = h1[row] . Wsbj[k] + b -------------
__global__ void k_emis(const f16* __restrict__ h1, const float* __restrict__ Wsbj,
                       const float* __restrict__ bsbj, float* __restrict__ em){
  int idx = blockIdx.x*blockDim.x + threadIdx.x;
  if (idx >= ROWS_*K_) return;
  int row = idx / K_, k = idx - row*K_;
  const f16* hr = h1 + (size_t)row * (2*H_);
  const float* wr = Wsbj + (size_t)k * (2*H_);
  float s = bsbj[k];
  for (int d = 0; d < 2*H_; d++) s += (float)hr[d] * wr[d];
  em[idx] = s;
}

// ---------------- CRF: one block per batch -----------------------------------
__global__ __launch_bounds__(64)
void k_crf(const float* __restrict__ em, const int* __restrict__ text,
           const int* __restrict__ sbj, const float* __restrict__ start_t,
           const float* __restrict__ end_t, const float* __restrict__ trans,
           float* __restrict__ llh, float* __restrict__ nvalid){
  int b = blockIdx.x;
  int k = threadIdx.x;
  __shared__ float sc[K_], nx[K_];
  if (k < K_) sc[k] = start_t[k] + em[(size_t)b*K_ + k];     // t = 0
  __syncthreads();
  for (int t = 1; t < T_; t++){
    int mt = text[b*T_ + t] != 0;
    if (k < K_){
      if (mt){
        float mx = -1e30f;
        for (int jj = 0; jj < K_; jj++) mx = fmaxf(mx, sc[jj] + trans[jj*K_ + k]);
        float s = 0.f;
        for (int jj = 0; jj < K_; jj++) s += __expf(sc[jj] + trans[jj*K_ + k] - mx);
        nx[k] = mx + __logf(s) + em[((size_t)t*B_ + b)*K_ + k];
      } else nx[k] = sc[k];
    }
    __syncthreads();
    if (k < K_) sc[k] = nx[k];
    __syncthreads();
  }
  if (k == 0){
    float mx = -1e30f;
    for (int jj = 0; jj < K_; jj++) mx = fmaxf(mx, sc[jj] + end_t[jj]);
    float s = 0.f;
    for (int jj = 0; jj < K_; jj++) s += __expf(sc[jj] + end_t[jj] - mx);
    float logZ = mx + __logf(s);
    int tg = sbj[b*T_];
    float num = start_t[tg] + em[(size_t)b*K_ + tg];
    int prev = tg, last = tg;
    int cnt = (text[b*T_] != 0) ? 1 : 0;
    for (int t = 1; t < T_; t++){
      int tgt = sbj[b*T_ + t];
      if (text[b*T_ + t] != 0){
        num += em[((size_t)t*B_ + b)*K_ + tgt] + trans[prev*K_ + tgt];
        cnt++; last = tgt;
      }
      prev = tgt;
    }
    num += end_t[last];
    llh[b]    = num - logZ;
    nvalid[b] = (float)cnt;
  }
}

__global__ void k_reduce(const float* __restrict__ llh, const float* __restrict__ nvalid,
                         float* __restrict__ out){
  if (threadIdx.x == 0 && blockIdx.x == 0){
    float sl = 0.f, sn = 0.f;
    for (int b = 0; b < B_; b++){ sl += llh[b]; sn += nvalid[b]; }
    out[0] = -(sl / sn);
  }
}

// ---------------------------------------------------------------------------
extern "C" void kernel_launch(void* const* d_in, const int* in_sizes, int n_in,
                              void* d_out, int out_size, void* d_ws, size_t ws_size,
                              hipStream_t stream){
  const int*   text  = (const int*)  d_in[0];
  const int*   sbj   = (const int*)  d_in[1];
  const float* emb   = (const float*)d_in[2];
  const float* Wih0f = (const float*)d_in[3];
  const float* Whh0f = (const float*)d_in[4];
  const float* b0f   = (const float*)d_in[5];
  const float* Wih0b = (const float*)d_in[6];
  const float* Whh0b = (const float*)d_in[7];
  const float* b0b   = (const float*)d_in[8];
  const float* Wih1f = (const float*)d_in[9];
  const float* Whh1f = (const float*)d_in[10];
  const float* b1f   = (const float*)d_in[11];
  const float* Wih1b = (const float*)d_in[12];
  const float* Whh1b = (const float*)d_in[13];
  const float* b1b   = (const float*)d_in[14];
  const float* Wsbj  = (const float*)d_in[15];
  const float* bsbj  = (const float*)d_in[16];
  const float* startt= (const float*)d_in[17];
  const float* endt  = (const float*)d_in[18];
  const float* trans = (const float*)d_in[19];

  char* ws = (char*)d_ws;
  size_t off = 0;
  auto carve = [&](size_t bytes)->char*{
    char* p = ws + off; off += (bytes + 255) & ~(size_t)255; return p; };

  f16*   xf    = (f16*)  carve((size_t)ROWS_*DP_*2);
  f16*   h0    = (f16*)  carve((size_t)ROWS_*512*2);
  f16*   h1    = (f16*)  carve((size_t)ROWS_*512*2);
  float* Gf    = (float*)carve((size_t)ROWS_*G4_*4);
  float* Gb    = (float*)carve((size_t)ROWS_*G4_*4);
  f16*   wi0f  = (f16*)  carve((size_t)G4_*DP_*2);
  f16*   wi0b  = (f16*)  carve((size_t)G4_*DP_*2);
  f16*   wi1f  = (f16*)  carve((size_t)G4_*512*2);
  f16*   wi1b  = (f16*)  carve((size_t)G4_*512*2);
  f16*   wh0f  = (f16*)  carve((size_t)G4_*H_*2);
  f16*   wh0b  = (f16*)  carve((size_t)G4_*H_*2);
  f16*   wh1f  = (f16*)  carve((size_t)G4_*H_*2);
  f16*   wh1b  = (f16*)  carve((size_t)G4_*H_*2);
  float* em    = (float*)carve((size_t)ROWS_*K_*4);
  float* llh   = (float*)carve(B_*4);
  float* nvld  = (float*)carve(B_*4);

  // weight conversion
  k_cvt<<<(G4_*DP_+255)/256, 256, 0, stream>>>(Wih0f, wi0f, G4_, D_,  DP_);
  k_cvt<<<(G4_*DP_+255)/256, 256, 0, stream>>>(Wih0b, wi0b, G4_, D_,  DP_);
  k_cvt<<<(G4_*512+255)/256, 256, 0, stream>>>(Wih1f, wi1f, G4_, 512, 512);
  k_cvt<<<(G4_*512+255)/256, 256, 0, stream>>>(Wih1b, wi1b, G4_, 512, 512);
  k_cvt<<<(G4_*H_+255)/256,  256, 0, stream>>>(Whh0f, wh0f, G4_, H_,  H_);
  k_cvt<<<(G4_*H_+255)/256,  256, 0, stream>>>(Whh0b, wh0b, G4_, H_,  H_);
  k_cvt<<<(G4_*H_+255)/256,  256, 0, stream>>>(Whh1f, wh1f, G4_, H_,  H_);
  k_cvt<<<(G4_*H_+255)/256,  256, 0, stream>>>(Whh1b, wh1b, G4_, H_,  H_);

  // embedding gather (t-major rows)
  k_gather<<<ROWS_, 256, 0, stream>>>(text, emb, xf);

  // layer 0: input gates (WMMA), then persistent recurrence (fwd+bwd concurrent)
  k_gemm_ih<<<2048, 256, 0, stream>>>(xf, wi0f, b0f, Gf, DP_);
  k_gemm_ih<<<2048, 256, 0, stream>>>(xf, wi0b, b0b, Gb, DP_);
  k_lstm<<<2, 512, 0, stream>>>(Gf, Gb, wh0f, wh0b, h0);

  // layer 1
  k_gemm_ih<<<2048, 256, 0, stream>>>(h0, wi1f, b1f, Gf, 512);
  k_gemm_ih<<<2048, 256, 0, stream>>>(h0, wi1b, b1b, Gb, 512);
  k_lstm<<<2, 512, 0, stream>>>(Gf, Gb, wh1f, wh1b, h1);

  // emissions + CRF + final scalar
  k_emis<<<(ROWS_*K_+255)/256, 256, 0, stream>>>(h1, Wsbj, bsbj, em);
  k_crf<<<B_, 64, 0, stream>>>(em, text, sbj, startt, endt, trans, llh, nvld);
  k_reduce<<<1, 64, 0, stream>>>(llh, nvld, (float*)d_out);
  (void)in_sizes; (void)n_in; (void)out_size; (void)ws_size;
}
